// emb_mlp_Layers_34479997452527
// MI455X (gfx1250) — compile-verified, matrix operand order checked by reading, not run on hardware
//
#include <hip/hip_runtime.h>
#include <hip/hip_bf16.h>
#include <math.h>

#define N_NODES 100000
#define N_REL   16
#define N_EDGES 1000000
#define D_IN    192
#define D_H     128
#define D_X     64
#define D_Y     32
#define D_O     16

typedef _Float16 v16h __attribute__((ext_vector_type(16)));
typedef _Float16 v8h  __attribute__((ext_vector_type(8)));
typedef float    v8f  __attribute__((ext_vector_type(8)));
typedef float    v4f  __attribute__((ext_vector_type(4)));

static __device__ __forceinline__ v16h cat8(v8h lo, v8h hi) {
    return __builtin_shufflevector(lo, hi, 0,1,2,3,4,5,6,7,8,9,10,11,12,13,14,15);
}

// ---------------------------------------------------------------------------
// Fused MLP:  x = (tanh(emb @ W1^T + b1)) @ W2^T + b2     [N,192]->[N,128]->[N,64]
// 256 threads = 8 waves; each wave owns a 16-row tile (128 rows / block).
// W1/W2 converted to f16 in LDS once per block; f32 accumulate via WMMA.
// Fragment layout (CDNA5 16-bit A/B, wave32): lane L holds row/col (L&15),
// k-chunks [half*8 .. +7] and [16+half*8 .. +23] per 32-wide k-step.
// ---------------------------------------------------------------------------
__launch_bounds__(256)
__global__ void mlp_wmma_kernel(const float* __restrict__ emb,
                                const float* __restrict__ w1, const float* __restrict__ b1,
                                const float* __restrict__ w2, const float* __restrict__ b2,
                                float* __restrict__ xout)
{
    __shared__ __align__(16) _Float16 sW1[D_H * D_IN];   // 48 KB; reused as H staging in phase 2
    __shared__ __align__(16) _Float16 sW2[D_X * D_H];    // 16 KB

    const int tid = threadIdx.x;
    for (int i = tid; i < D_H * D_IN; i += 256) sW1[i] = (_Float16)w1[i];
    for (int i = tid; i < D_X * D_H;  i += 256) sW2[i] = (_Float16)w2[i];
    __syncthreads();

    const int wave = tid >> 5;
    const int lane = tid & 31;
    const int half = lane >> 4;
    const int m    = lane & 15;
    const int rowBase = blockIdx.x * 128 + wave * 16;

    // ---------- phase 1: H = tanh(emb @ W1^T + b1), 16x128 per wave ----------
    v8f accH[8] = {};
    const int gr = min(rowBase + m, N_NODES - 1);            // clamped load row
    const float* arow = emb + (size_t)gr * D_IN + half * 8;

    for (int ks = 0; ks < 6; ++ks) {                         // K = 192 = 6 * 32
        const float* ap = arow + ks * 32;
        v4f f0 = *(const v4f*)(ap);
        v4f f1 = *(const v4f*)(ap + 4);
        v4f f2 = *(const v4f*)(ap + 16);
        v4f f3 = *(const v4f*)(ap + 20);
        v8h lo, hi;
#pragma unroll
        for (int i = 0; i < 4; ++i) {
            lo[i]     = (_Float16)f0[i];  lo[i + 4] = (_Float16)f1[i];
            hi[i]     = (_Float16)f2[i];  hi[i + 4] = (_Float16)f3[i];
        }
        v16h a = cat8(lo, hi);
#pragma unroll
        for (int nt = 0; nt < 8; ++nt) {                     // 128 cols = 8 n-tiles
            const _Float16* wp = sW1 + (nt * 16 + m) * D_IN + ks * 32 + half * 8;
            v16h b = cat8(*(const v8h*)wp, *(const v8h*)(wp + 16));
            accH[nt] = __builtin_amdgcn_wmma_f32_16x16x32_f16(
                false, a, false, b, (short)0, accH[nt], false, false);
        }
    }
    __syncthreads();  // all waves done reading sW1 before it is overwritten by H

    // stage H (f16) back to LDS in row-major A layout (private per-wave region)
    _Float16* Hs = sW1 + wave * (16 * D_H);
#pragma unroll
    for (int nt = 0; nt < 8; ++nt) {
        const int  col  = nt * 16 + m;
        const float bias = b1[col];
#pragma unroll
        for (int r = 0; r < 8; ++r) {
            const int mrow = half * 8 + r;                   // D layout: row = 8*half + r
            Hs[mrow * D_H + col] = (_Float16)tanhf(accH[nt][r] + bias);
        }
    }

    // ---------- phase 2: X = H @ W2^T + b2, 16x64 per wave ----------
    v8f accX[4] = {};
    for (int ks = 0; ks < 4; ++ks) {                         // K = 128 = 4 * 32
        const _Float16* hp = Hs + m * D_H + ks * 32 + half * 8;
        v16h a = cat8(*(const v8h*)hp, *(const v8h*)(hp + 16));
#pragma unroll
        for (int nt = 0; nt < 4; ++nt) {                     // 64 cols = 4 n-tiles
            const _Float16* wp = sW2 + (nt * 16 + m) * D_H + ks * 32 + half * 8;
            v16h b = cat8(*(const v8h*)wp, *(const v8h*)(wp + 16));
            accX[nt] = __builtin_amdgcn_wmma_f32_16x16x32_f16(
                false, a, false, b, (short)0, accX[nt], false, false);
        }
    }
#pragma unroll
    for (int nt = 0; nt < 4; ++nt) {
        const int  col  = nt * 16 + m;
        const float bias = b2[col];
#pragma unroll
        for (int r = 0; r < 8; ++r) {
            const int row = rowBase + half * 8 + r;
            if (row < N_NODES)
                xout[(size_t)row * D_X + col] = accX[nt][r] + bias;
        }
    }
}

// ---------------------------------------------------------------------------
// Per-(dst,rel) edge counts (shared by both RGCN layers: seg = dst*R + etype)
// ---------------------------------------------------------------------------
__global__ void cnt_kernel(const int* __restrict__ ei, const int* __restrict__ et,
                           float* __restrict__ cnt)
{
    int e = blockIdx.x * blockDim.x + threadIdx.x;
    if (e < N_EDGES) {
        int dst = ei[N_EDGES + e];
        int r   = et[e];
        atomicAdd(&cnt[dst * N_REL + r], 1.0f);
    }
}

// acc1[n] = x[n] @ root1 + b1   (32 lanes = 32 output cols per node)
__global__ void root1_kernel(const float* __restrict__ x, const float* __restrict__ root,
                             const float* __restrict__ bias, float* __restrict__ acc)
{
    int t = blockIdx.x * blockDim.x + threadIdx.x;
    int node = t >> 5, col = t & 31;
    if (node >= N_NODES) return;
    float s = bias[col];
    const float* xr = x + (size_t)node * D_X;
#pragma unroll 8
    for (int f = 0; f < D_X; ++f) s += xr[f] * root[f * D_Y + col];
    acc[(size_t)node * D_Y + col] = s;
}

// per-edge: acc1[dst] += (x[src] @ W_{etype}) / cnt[dst,etype]   (one wave per edge)
__global__ void edge1_kernel(const int* __restrict__ ei, const int* __restrict__ et,
                             const float* __restrict__ cnt, const float* __restrict__ x,
                             const float* __restrict__ w, float* __restrict__ acc)
{
    int t = blockIdx.x * blockDim.x + threadIdx.x;
    int e = t >> 5, col = t & 31;
    if (e >= N_EDGES) return;
    int src = ei[e], dst = ei[N_EDGES + e], r = et[e];
    float scale = 1.0f / fmaxf(cnt[dst * N_REL + r], 1.0f);
    const float* xr = x + (size_t)src * D_X;
    const float* wr = w + (size_t)r * D_X * D_Y + col;       // w1[r][f][col]
    float s = 0.0f;
#pragma unroll 8
    for (int f = 0; f < D_X; ++f) s += xr[f] * wr[f * D_Y];
    atomicAdd(&acc[(size_t)dst * D_Y + col], s * scale);
}

__global__ void relu_kernel(float* __restrict__ a, int n)
{
    int i = blockIdx.x * blockDim.x + threadIdx.x;
    if (i < n) a[i] = fmaxf(a[i], 0.0f);
}

// out[n] = y[n] @ root2 + b2   (16 lanes per node)
__global__ void root2_kernel(const float* __restrict__ y, const float* __restrict__ root,
                             const float* __restrict__ bias, float* __restrict__ out)
{
    int t = blockIdx.x * blockDim.x + threadIdx.x;
    int node = t >> 4, col = t & 15;
    if (node >= N_NODES) return;
    float s = bias[col];
    const float* yr = y + (size_t)node * D_Y;
#pragma unroll 8
    for (int f = 0; f < D_Y; ++f) s += yr[f] * root[f * D_O + col];
    out[(size_t)node * D_O + col] = s;
}

// per-edge: out[dst] += (y[src] @ W2_{etype}) / cnt[dst,etype]   (16 lanes per edge)
__global__ void edge2_kernel(const int* __restrict__ ei, const int* __restrict__ et,
                             const float* __restrict__ cnt, const float* __restrict__ y,
                             const float* __restrict__ w, float* __restrict__ out)
{
    int t = blockIdx.x * blockDim.x + threadIdx.x;
    int e = t >> 4, col = t & 15;
    if (e >= N_EDGES) return;
    int src = ei[e], dst = ei[N_EDGES + e], r = et[e];
    float scale = 1.0f / fmaxf(cnt[dst * N_REL + r], 1.0f);
    const float* yr = y + (size_t)src * D_Y;
    const float* wr = w + (size_t)r * D_Y * D_O + col;       // w2[r][f][col]
    float s = 0.0f;
#pragma unroll 8
    for (int f = 0; f < D_Y; ++f) s += yr[f] * wr[f * D_O];
    atomicAdd(&out[(size_t)dst * D_O + col], s * scale);
}

__global__ void sigmoid_kernel(float* __restrict__ a, int n)
{
    int i = blockIdx.x * blockDim.x + threadIdx.x;
    if (i < n) a[i] = 1.0f / (1.0f + expf(-a[i]));
}

// ---------------------------------------------------------------------------
extern "C" void kernel_launch(void* const* d_in, const int* in_sizes, int n_in,
                              void* d_out, int out_size, void* d_ws, size_t ws_size,
                              hipStream_t stream)
{
    const float* emb   = (const float*)d_in[0];
    const float* l1w   = (const float*)d_in[1];
    const float* l1b   = (const float*)d_in[2];
    const float* l2w   = (const float*)d_in[3];
    const float* l2b   = (const float*)d_in[4];
    const float* w1    = (const float*)d_in[5];
    const float* root1 = (const float*)d_in[6];
    const float* b1    = (const float*)d_in[7];
    const float* w2    = (const float*)d_in[8];
    const float* root2 = (const float*)d_in[9];
    const float* b2    = (const float*)d_in[10];
    const int*   ei    = (const int*)d_in[11];   // [2, E]
    const int*   et    = (const int*)d_in[12];   // [E]
    float*       out   = (float*)d_out;          // [N, 16] (used as acc2 in place)

    char*  ws   = (char*)d_ws;
    float* x    = (float*)(ws);                                   // N*64 f32 (25.6 MB)
    float* cnt  = (float*)(ws + (size_t)N_NODES * 64 * 4);        // N*16 f32 ( 6.4 MB)
    float* acc1 = (float*)(ws + (size_t)N_NODES * 80 * 4);        // N*32 f32 (12.8 MB)

    hipMemsetAsync(cnt, 0, (size_t)N_NODES * N_REL * sizeof(float), stream);

    // dense MLP (WMMA)
    mlp_wmma_kernel<<<(N_NODES + 127) / 128, 256, 0, stream>>>(emb, l1w, l1b, l2w, l2b, x);

    // segment counts (shared by both layers)
    cnt_kernel<<<(N_EDGES + 255) / 256, 256, 0, stream>>>(ei, et, cnt);

    // RGCN layer 1: acc1 = x@root1 + b1, then scatter edge messages, then ReLU
    root1_kernel<<<(N_NODES * 32 + 255) / 256, 256, 0, stream>>>(x, root1, b1, acc1);
    edge1_kernel<<<(N_EDGES * 32 + 255) / 256, 256, 0, stream>>>(ei, et, cnt, x, w1, acc1);
    relu_kernel<<<(N_NODES * 32 + 255) / 256, 256, 0, stream>>>(acc1, N_NODES * 32);

    // RGCN layer 2 (accumulate directly into d_out), then sigmoid in place
    root2_kernel<<<(N_NODES * 16 + 255) / 256, 256, 0, stream>>>(acc1, root2, b2, out);
    edge2_kernel<<<(N_EDGES * 16 + 255) / 256, 256, 0, stream>>>(ei, et, cnt, acc1, w2, out);
    sigmoid_kernel<<<(N_NODES * 16 + 255) / 256, 256, 0, stream>>>(out, N_NODES * 16);
}
